// JanetLayer_88175678587048
// MI455X (gfx1250) — matrix operand order repaired and checked by reading, not compiled, not run on hardware
//
#include <hip/hip_runtime.h>
#include <hip/hip_bf16.h>

// ---------------------------------------------------------------------------
// JANET layer for MI455X (gfx1250): bf16 WMMA GEMMs, LDS-resident recurrence.
// T=2048, B=64, I=H=256.
// Key insight: the recurrence never mixes batch rows, so each 16-row batch
// slice runs on its own WGP with h in LDS and s_barrier-only sync.
// ---------------------------------------------------------------------------

typedef float  v8f   __attribute__((ext_vector_type(8)));
typedef __bf16 v16bf __attribute__((ext_vector_type(16)));

union Frag {
    unsigned u[8];
    uint4    u4[2];
    v16bf    v;
};

#define T_STEPS 2048
#define BATCH   64
#define IDIM    256
#define HDIM    256

// ws layout (bytes)
#define OFF_PK   256
#define PKB      131072                          // one packed 256x256 bf16 matrix
#define OFF_XF   (OFF_PK + 4*PKB)                // 524544
#define XFB      ((size_t)T_STEPS*BATCH*HDIM*4)  // 134217728
#define OFF_XH   (OFF_XF + XFB)

// ---- helpers --------------------------------------------------------------

__device__ __forceinline__ unsigned pk_bf16(float lo, float hi) {
    // round-to-nearest-even f32 -> bf16, packed pair (low = first element)
    unsigned a = __float_as_uint(lo);
    unsigned b = __float_as_uint(hi);
    a += 0x7fffu + ((a >> 16) & 1u);
    b += 0x7fffu + ((b >> 16) & 1u);
    return (a >> 16) | (b & 0xffff0000u);
}

__device__ __forceinline__ unsigned short bf16_1(float x) {
    unsigned u = __float_as_uint(x);
    u += 0x7fffu + ((u >> 16) & 1u);
    return (unsigned short)(u >> 16);
}

__device__ __forceinline__ float fast_sigmoid(float x) {
    return 1.0f / (1.0f + __expf(-x));
}
__device__ __forceinline__ float fast_tanh(float x) {
    float e = __expf(-2.0f * x);
    return (1.0f - e) / (1.0f + e);
}

// ---- kernel 1: pack weights into B-fragment-native bf16 layout ------------
// B fragment (32x16, KxN) per ISA 7.12.2: lane L -> N = L%16,
// K = (v/4)*16 + (L/16)*8 + 2(v%4) (+pair), v in [0,8). Packed dword index:
//   ((nt*8 + kt)*32 + lane)*8 + v
__global__ void __launch_bounds__(256)
janet_pack_kernel(const float* __restrict__ Wf, const float* __restrict__ Wh,
                  const float* __restrict__ Uf, const float* __restrict__ Uh,
                  unsigned* __restrict__ wfp, unsigned* __restrict__ whp,
                  unsigned* __restrict__ ufp, unsigned* __restrict__ uhp) {
    int tid = blockIdx.x * 256 + threadIdx.x;   // 131072 threads total
    int mat  = tid >> 15;          // 0:Wf 1:Wh 2:Uf 3:Uh
    int r    = tid & 32767;
    int v    = r & 7;
    int lane = (r >> 3) & 31;
    int frag = r >> 8;             // = nt*8 + kt
    int kt   = frag & 7;
    int nt   = frag >> 3;
    int N = nt * 16 + (lane & 15);
    int K = kt * 32 + (v >> 2) * 16 + (lane >> 4) * 8 + 2 * (v & 3);
    const float* src = (mat == 0) ? Wf : (mat == 1) ? Wh : (mat == 2) ? Uf : Uh;
    unsigned*    dst = (mat == 0) ? wfp : (mat == 1) ? whp : (mat == 2) ? ufp : uhp;
    dst[r] = pk_bf16(src[K * HDIM + N], src[(K + 1) * HDIM + N]);
}

// ---- kernel 2: xf = seq@W_f + b_f ; xh = seq@W_h + b_h --------------------
// One 16-row M-tile of [T*B, I] per block; 8 waves: 4 per gate, 4 N-tiles
// per wave. Activation tile staged once in LDS as packed bf16.
__global__ void __launch_bounds__(256)
janet_xproj_kernel(const float* __restrict__ seq,
                   const unsigned* __restrict__ wf_pk, const unsigned* __restrict__ wh_pk,
                   const float* __restrict__ b_f, const float* __restrict__ b_h,
                   float* __restrict__ xf, float* __restrict__ xh) {
    __shared__ unsigned ldsA[16 * 128];          // 16 rows x 256 bf16 (8 KB)
    const int row0 = blockIdx.x * 16;            // row in [T*B]
    const int tid  = threadIdx.x;

    // cooperative stage: f32 -> packed bf16 pairs
    #pragma unroll
    for (int i = 0; i < 8; ++i) {
        int p  = tid * 8 + i;                    // pair index 0..2047
        int r  = p >> 7;
        int kp = p & 127;
        const float2 f2 = *(const float2*)(seq + (size_t)(row0 + r) * IDIM + kp * 2);
        ldsA[p] = pk_bf16(f2.x, f2.y);
    }
    __syncthreads();

    const int lane = tid & 31;
    const int w    = tid >> 5;
    const int gate = w >> 2;                     // 0 -> f, 1 -> h
    const int ng0  = (w & 3) * 4;                // first of 4 N-tiles
    const unsigned* pk = gate ? wh_pk : wf_pk;

    const int arow = lane & 15;
    const int quad = lane >> 4;

    v8f acc[4] = {{}, {}, {}, {}};
    for (int kt = 0; kt < 8; ++kt) {
        Frag a;
        int base = arow * 128 + kt * 16 + quad * 4;
        #pragma unroll
        for (int v = 0; v < 4; ++v) {
            a.u[v]     = ldsA[base + v];
            a.u[4 + v] = ldsA[base + 8 + v];
        }
        #pragma unroll
        for (int j = 0; j < 4; ++j) {
            Frag b;
            const uint4* bp = (const uint4*)(pk + (size_t)(((ng0 + j) * 8 + kt) * 32 + lane) * 8);
            b.u4[0] = bp[0];
            b.u4[1] = bp[1];
            acc[j] = __builtin_amdgcn_wmma_f32_16x16x32_bf16(
                false, a.v, false, b.v, (short)0, acc[j], false, false);
        }
    }

    const float* bias = gate ? b_h : b_f;
    float*       dst  = gate ? xh : xf;
    #pragma unroll
    for (int j = 0; j < 4; ++j) {
        int col = (ng0 + j) * 16 + (lane & 15);
        float bv = bias[col];
        #pragma unroll
        for (int v = 0; v < 8; ++v) {
            int mr = quad * 8 + v;               // C layout: M = (lane/16)*8 + v
            dst[(size_t)(row0 + mr) * HDIM + col] = acc[j][v] + bv;
        }
    }
}

// ---- kernel 3: LDS-resident sequential recurrence -------------------------
// 4 WGs (one per 16-row batch slice), 512 threads = 16 waves; wave nt owns
// C-tile cols [16nt,16nt+16) for BOTH gates (16 WMMAs/step). h ping-pongs in
// LDS (bf16 A-layout + f32 copy); U_f/U_h fragments live in VGPRs for the
// whole 2047-step loop; per-step sync is a plain workgroup barrier.
__global__ void __launch_bounds__(512)
janet_recur_kernel(const unsigned* __restrict__ uf_pk, const unsigned* __restrict__ uh_pk,
                   const float* __restrict__ xf, const float* __restrict__ xh,
                   const float* __restrict__ prelu_a,
                   float* __restrict__ out) {
    __shared__ unsigned short hbf[2][16 * 256];  // bf16 h ping-pong (16 KB)
    __shared__ float          hfp[2][16 * 256];  // f32  h ping-pong (32 KB)

    const int tid  = threadIdx.x;
    const int m    = blockIdx.x;                 // batch row-group, 0..3
    const float pa = prelu_a[0];

    const int lane = tid & 31;
    const int nt   = tid >> 5;                   // wave = H col-tile, 0..15
    const int quad = lane >> 4;
    const int col  = nt * 16 + (lane & 15);      // C column
    const int arow = lane & 15;                  // local A-fragment row

    // Hoist U fragments into registers: 2 gates x 8 kt x 8 dwords = 128 VGPRs
    Frag bF[8], bH[8];
    #pragma unroll
    for (int kt = 0; kt < 8; ++kt) {
        size_t boff = (size_t)((nt * 8 + kt) * 32 + lane) * 8;
        bF[kt].u4[0] = ((const uint4*)(uf_pk + boff))[0];
        bF[kt].u4[1] = ((const uint4*)(uf_pk + boff))[1];
        bH[kt].u4[0] = ((const uint4*)(uh_pk + boff))[0];
        bH[kt].u4[1] = ((const uint4*)(uh_pk + boff))[1];
    }

    // t = 0: h0 = prelu((1 - sigmoid(xf0)) * tanh(xh0)) for this batch slice
    #pragma unroll
    for (int i = 0; i < 8; ++i) {
        int idx  = tid * 8 + i;                  // 0..4095 (16x256 local)
        int gidx = (m * 16 + (idx >> 8)) * 256 + (idx & 255);
        float f0 = fast_sigmoid(xf[gidx]);
        float h  = (1.0f - f0) * fast_tanh(xh[gidx]);
        h = (h >= 0.0f) ? h : pa * h;
        hfp[0][idx] = h;
        hbf[0][idx] = bf16_1(h);
        out[gidx]   = h;
    }
    __syncthreads();

    for (int t = 1; t < T_STEPS; ++t) {
        const int pp = (t - 1) & 1;              // read buffer
        const int pc = t & 1;                    // write buffer
        const size_t tb = ((size_t)t * BATCH + m * 16) * 256;

        // Early-issue the h-independent loads; they overlap the WMMA chain.
        float xfv[8], xhv[8], hpv[8];
        #pragma unroll
        for (int v = 0; v < 8; ++v) {
            int lrow = quad * 8 + v;             // C layout local row
            xfv[v] = xf[tb + (size_t)lrow * 256 + col];
            xhv[v] = xh[tb + (size_t)lrow * 256 + col];
            hpv[v] = hfp[pp][lrow * 256 + col];
        }

        v8f accF = {};
        v8f accH = {};
        #pragma unroll
        for (int kt = 0; kt < 8; ++kt) {
            Frag a;
            const uint4* p = (const uint4*)&hbf[pp][arow * 256 + kt * 32 + quad * 8];
            a.u4[0] = p[0];                      // K offsets [quad*8, quad*8+8)
            a.u4[1] = p[2];                      // K offsets [16+quad*8, ...)
            accF = __builtin_amdgcn_wmma_f32_16x16x32_bf16(
                false, a.v, false, bF[kt].v, (short)0, accF, false, false);
            accH = __builtin_amdgcn_wmma_f32_16x16x32_bf16(
                false, a.v, false, bH[kt].v, (short)0, accH, false, false);
        }

        #pragma unroll
        for (int v = 0; v < 8; ++v) {
            int lrow = quad * 8 + v;
            float fg = fast_sigmoid(accF[v] + xfv[v]);
            float cd = fast_tanh(accH[v] + xhv[v]);
            float h  = fg * hpv[v] + (1.0f - fg) * cd;
            h = (h >= 0.0f) ? h : pa * h;
            hfp[pc][lrow * 256 + col] = h;
            hbf[pc][lrow * 256 + col] = bf16_1(h);
            out[tb + (size_t)lrow * 256 + col] = h;
        }
        __syncthreads();
    }
}

// ---- launch ---------------------------------------------------------------

extern "C" void kernel_launch(void* const* d_in, const int* in_sizes, int n_in,
                              void* d_out, int out_size, void* d_ws, size_t ws_size,
                              hipStream_t stream) {
    (void)in_sizes; (void)n_in; (void)out_size; (void)ws_size;

    const float* seq = (const float*)d_in[0];   // [T,B,I]
    const float* Wf  = (const float*)d_in[1];   // [I,H]
    const float* Uf  = (const float*)d_in[2];   // [H,H]
    const float* bf_ = (const float*)d_in[3];   // [H]
    const float* Wh  = (const float*)d_in[4];   // [I,H]
    const float* Uh  = (const float*)d_in[5];   // [H,H]
    const float* bh_ = (const float*)d_in[6];   // [H]
    const float* pa  = (const float*)d_in[7];   // [1]

    char* ws = (char*)d_ws;
    unsigned* wf_pk = (unsigned*)(ws + OFF_PK + 0 * PKB);
    unsigned* wh_pk = (unsigned*)(ws + OFF_PK + 1 * PKB);
    unsigned* uf_pk = (unsigned*)(ws + OFF_PK + 2 * PKB);
    unsigned* uh_pk = (unsigned*)(ws + OFF_PK + 3 * PKB);
    float*    xf    = (float*)(ws + OFF_XF);
    float*    xh    = (float*)(ws + OFF_XH);

    // 1) pack weights to bf16 B-fragment layout
    janet_pack_kernel<<<512, 256, 0, stream>>>(Wf, Wh, Uf, Uh,
                                               wf_pk, wh_pk, uf_pk, uh_pk);
    // 2) input projections (bandwidth-bound WMMA GEMM)
    janet_xproj_kernel<<<(T_STEPS * BATCH) / 16, 256, 0, stream>>>(
        seq, wf_pk, wh_pk, bf_, bh_, xf, xh);
    // 3) per-batch-slice LDS-resident recurrence (4 WGPs, s_barrier sync only)
    janet_recur_kernel<<<4, 512, 0, stream>>>(uf_pk, uh_pk, xf, xh, pa,
                                              (float*)d_out);
}
